// CrossAttentionSubLayer_18794776887528
// MI455X (gfx1250) — compile-verified, matrix operand order checked by reading, not run on hardware
//
#include <hip/hip_runtime.h>

typedef __attribute__((ext_vector_type(16))) __bf16 v16bf;
typedef __attribute__((ext_vector_type(8)))  float  v8f;

#define NEG_MASK (-1e13f)

__device__ __forceinline__ unsigned short f2bf(float f) {
  unsigned int u = __float_as_uint(f);
  u += 0x7fffu + ((u >> 16) & 1u);          // round-to-nearest-even
  return (unsigned short)(u >> 16);
}
__device__ __forceinline__ float bf2f(unsigned short h) {
  return __uint_as_float(((unsigned int)h) << 16);
}

// ---------------------------------------------------------------------------
// Vectorized f32 -> bf16 conversion (4 elems/thread)
// ---------------------------------------------------------------------------
__global__ __launch_bounds__(256)
void convert_f32_bf16_v4(const float4* __restrict__ src,
                         ushort4* __restrict__ dst, int n4) {
  int i = blockIdx.x * 256 + threadIdx.x;
  if (i < n4) {
    float4 f = src[i];
    ushort4 o;
    o.x = f2bf(f.x); o.y = f2bf(f.y); o.z = f2bf(f.z); o.w = f2bf(f.w);
    dst[i] = o;
  }
}

// ---------------------------------------------------------------------------
// Per-batch transpose + convert: Vt[b][d][k] = (bf16)V[b][k][d]
// ---------------------------------------------------------------------------
__global__ __launch_bounds__(256)
void transpose_convert_v(const float* __restrict__ V,
                         unsigned short* __restrict__ Vt, int L, int D) {
  __shared__ float tile[32][33];
  const int b  = blockIdx.z;
  const int d0 = blockIdx.x * 32;
  const int k0 = blockIdx.y * 32;
  const float* Vb = V + (size_t)b * L * D;
  unsigned short* Vtb = Vt + (size_t)b * L * D;
  for (int i = threadIdx.y; i < 32; i += 8)
    tile[i][threadIdx.x] = Vb[(size_t)(k0 + i) * D + d0 + threadIdx.x];
  __syncthreads();
  for (int i = threadIdx.y; i < 32; i += 8)
    Vtb[(size_t)(d0 + i) * L + k0 + threadIdx.x] = f2bf(tile[threadIdx.x][i]);
}

// ---------------------------------------------------------------------------
// Generic bf16 WMMA GEMM:  C[M,N] = A[M,K] * Brows[N,K]^T
//   MODE 0: C = scale*A*B^T, pad-mask rectangle to NEG, store bf16  (scores)
//   MODE 1: C = A*B^T, store bf16                                   (attn @ V)
//   MODE 2: C = A*B^T + bias[col], store f32                        (final proj)
// 128x128 tile / workgroup, BLK_K=32, 8 waves (4 M x 2 N), each wave 32x64.
// ---------------------------------------------------------------------------
union FragCast { uint4 u[2]; v16bf v; };

template <int MODE>
__global__ __launch_bounds__(256)
void gemm_wmma(const unsigned short* __restrict__ A,
               const unsigned short* __restrict__ Bm,
               void* __restrict__ Cout,
               int K, int ldc,
               long long sA, long long sB, long long sC,
               float scale,
               const int* __restrict__ qpad, const int* __restrict__ kpad,
               const float* __restrict__ bias) {
  __shared__ unsigned short shA[2][128 * 32];
  __shared__ unsigned short shB[2][128 * 32];

  const int b = blockIdx.z;
  const unsigned short* Ab = A  + (size_t)b * sA;
  const unsigned short* Bb = Bm + (size_t)b * sB;
  const int row0 = blockIdx.y * 128;
  const int col0 = blockIdx.x * 128;

  const int t    = threadIdx.x;
  const int lane = t & 31;
  const int wave = t >> 5;
  const int wm   = wave >> 1;   // 0..3 : 32 rows each
  const int wn   = wave & 1;    // 0..1 : 64 cols each
  const int lh   = lane & 15;
  const int hsel = lane >> 4;   // 0/1 (wave32 half select per ISA 16-bit layout)

  v8f acc[2][4];
#pragma unroll
  for (int i = 0; i < 2; i++)
#pragma unroll
    for (int j = 0; j < 4; j++) {
      v8f z = {0.f, 0.f, 0.f, 0.f, 0.f, 0.f, 0.f, 0.f};
      acc[i][j] = z;
    }

  // Staging: 128 rows x 32 bf16 tile; 4 threads/row (16B each), rows sr, sr+64.
  const int sr = t >> 2;
  const int ss = t & 3;
  const size_t rowStep = (size_t)64 * K;      // elements to row sr+64
  const unsigned short* gA = Ab + (size_t)(row0 + sr) * K + ss * 8;
  const unsigned short* gB = Bb + (size_t)(col0 + sr) * K + ss * 8;
  const int woff = sr * 32 + ss * 8;          // LDS write offset (elements)

  // Prologue: tile 0 -> buffer 0
  uint4 ra0 = *(const uint4*)(gA);
  uint4 ra1 = *(const uint4*)(gA + rowStep);
  uint4 rb0 = *(const uint4*)(gB);
  uint4 rb1 = *(const uint4*)(gB + rowStep);
  *(uint4*)(&shA[0][woff])        = ra0;
  *(uint4*)(&shA[0][woff + 2048]) = ra1;
  *(uint4*)(&shB[0][woff])        = rb0;
  *(uint4*)(&shB[0][woff + 2048]) = rb1;
  __syncthreads();

  // Per-wave LDS fragment base offsets (elements), hoisted out of the K loop.
  int aBase[2], bBase[4];
#pragma unroll
  for (int sm = 0; sm < 2; sm++)
    aBase[sm] = (wm * 32 + sm * 16 + lh) * 32 + hsel * 8;
#pragma unroll
  for (int sn = 0; sn < 4; sn++)
    bBase[sn] = (wn * 64 + sn * 16 + lh) * 32 + hsel * 8;

  const int KT = K >> 5;
  for (int kt = 0; kt < KT; ++kt) {
    const int cur = kt & 1;
    const bool pre = (kt + 1) < KT;
    if (pre) {                                 // fetch next tile to registers
      gA += 32; gB += 32;
      ra0 = *(const uint4*)(gA);
      ra1 = *(const uint4*)(gA + rowStep);
      rb0 = *(const uint4*)(gB);
      rb1 = *(const uint4*)(gB + rowStep);
    }
    if (kt + 2 < KT) {                         // global_prefetch_b8: tile after next
      __builtin_prefetch(gA + 32, 0, 1);
      __builtin_prefetch(gB + 32, 0, 1);
    }

    // Fragments: 16-bit 16x32 layout (lanes 0-15 K{0..7,16..23},
    // lanes 16-31 K{8..15,24..31}); two 16B LDS reads each.
    const unsigned short* sAc = &shA[cur][0];
    const unsigned short* sBc = &shB[cur][0];
    v16bf afrag[2], bfrag[4];
#pragma unroll
    for (int sm = 0; sm < 2; sm++) {
      const unsigned short* p = sAc + aBase[sm];
      FragCast fc;
      fc.u[0] = *(const uint4*)p;
      fc.u[1] = *(const uint4*)(p + 16);
      afrag[sm] = fc.v;
    }
#pragma unroll
    for (int sn = 0; sn < 4; sn++) {
      const unsigned short* p = sBc + bBase[sn];
      FragCast fc;
      fc.u[0] = *(const uint4*)p;
      fc.u[1] = *(const uint4*)(p + 16);
      bfrag[sn] = fc.v;
    }

#pragma unroll
    for (int sm = 0; sm < 2; sm++)
#pragma unroll
      for (int sn = 0; sn < 4; sn++)
        acc[sm][sn] = __builtin_amdgcn_wmma_f32_16x16x32_bf16(
            false, afrag[sm], false, bfrag[sn], (short)0, acc[sm][sn],
            false, false);

    if (pre) {                                 // stash next tile in other buffer
      unsigned short* dA = &shA[cur ^ 1][0];
      unsigned short* dB = &shB[cur ^ 1][0];
      *(uint4*)(dA + woff)        = ra0;
      *(uint4*)(dA + woff + 2048) = ra1;
      *(uint4*)(dB + woff)        = rb0;
      *(uint4*)(dB + woff + 2048) = rb1;
    }
    __syncthreads();
  }

  // C layout (ISA 7.12.2): VGPR i -> M = i + 8*hsel, N = lane%16
  int qp = 0, kp = 0;
  if (MODE == 0) { qp = qpad[b]; kp = kpad[b]; }
#pragma unroll
  for (int sm = 0; sm < 2; sm++) {
#pragma unroll
    for (int sn = 0; sn < 4; sn++) {
#pragma unroll
      for (int i = 0; i < 8; i++) {
        float v = acc[sm][sn][i];
        const int r = row0 + wm * 32 + sm * 16 + hsel * 8 + i;
        const int c = col0 + wn * 64 + sn * 16 + lh;
        if (MODE == 0) {
          v *= scale;
          if (r >= qp && c >= kp) v = NEG_MASK;
          ((unsigned short*)Cout)[(size_t)b * sC + (size_t)r * ldc + c] = f2bf(v);
        } else if (MODE == 1) {
          ((unsigned short*)Cout)[(size_t)b * sC + (size_t)r * ldc + c] = f2bf(v);
        } else {
          ((float*)Cout)[(size_t)r * ldc + c] = v + bias[c];
        }
      }
    }
  }
}

// ---------------------------------------------------------------------------
// In-place row softmax over bf16 scores: one workgroup per row (2048 cols)
// ---------------------------------------------------------------------------
__global__ __launch_bounds__(256)
void softmax_rows(unsigned short* __restrict__ S, int ncols) {
  __shared__ float red[8];
  const size_t row = blockIdx.x;
  unsigned short* p = S + row * (size_t)ncols;
  const int t = threadIdx.x, lane = t & 31, wave = t >> 5;

  float v[8];
  float mx = -3.0e38f;
#pragma unroll
  for (int j = 0; j < 8; j++) {
    v[j] = bf2f(p[t + j * 256]);
    mx = fmaxf(mx, v[j]);
  }
  for (int o = 16; o; o >>= 1) mx = fmaxf(mx, __shfl_xor(mx, o, 32));
  if (lane == 0) red[wave] = mx;
  __syncthreads();
  mx = red[0];
#pragma unroll
  for (int w = 1; w < 8; w++) mx = fmaxf(mx, red[w]);
  __syncthreads();

  float sum = 0.f;
#pragma unroll
  for (int j = 0; j < 8; j++) {
    v[j] = __expf(v[j] - mx);
    sum += v[j];
  }
  for (int o = 16; o; o >>= 1) sum += __shfl_xor(sum, o, 32);
  if (lane == 0) red[wave] = sum;
  __syncthreads();
  sum = 0.f;
#pragma unroll
  for (int w = 0; w < 8; w++) sum += red[w];
  const float inv = 1.0f / sum;
#pragma unroll
  for (int j = 0; j < 8; j++) p[t + j * 256] = f2bf(v[j] * inv);
}

// ---------------------------------------------------------------------------
// Launcher
// ---------------------------------------------------------------------------
extern "C" void kernel_launch(void* const* d_in, const int* in_sizes, int n_in,
                              void* d_out, int out_size, void* d_ws,
                              size_t ws_size, hipStream_t stream) {
  constexpr int B = 8, L = 2048, D = 1024, E = 1024;
  const float* Q    = (const float*)d_in[0];
  const float* K    = (const float*)d_in[1];
  const float* V    = (const float*)d_in[2];
  const int*   qpad = (const int*)d_in[3];
  const int*   kpad = (const int*)d_in[4];
  const float* W    = (const float*)d_in[5];
  const float* bias = (const float*)d_in[6];

  const size_t nQKV = (size_t)B * L * D;  // 16,777,216
  const size_t nW   = (size_t)E * D;      //  1,048,576
  const size_t nS   = (size_t)B * L * L;  // 33,554,432

  unsigned short* Qb = (unsigned short*)d_ws;
  unsigned short* Kb = Qb + nQKV;
  unsigned short* Vt = Kb + nQKV;
  unsigned short* Wb = Vt + nQKV;
  unsigned short* S  = Wb + nW;
  unsigned short* Ob = S + nS;

  convert_f32_bf16_v4<<<(int)(nQKV / 4 / 256), 256, 0, stream>>>(
      (const float4*)Q, (ushort4*)Qb, (int)(nQKV / 4));
  convert_f32_bf16_v4<<<(int)(nQKV / 4 / 256), 256, 0, stream>>>(
      (const float4*)K, (ushort4*)Kb, (int)(nQKV / 4));
  convert_f32_bf16_v4<<<(int)(nW / 4 / 256), 256, 0, stream>>>(
      (const float4*)W, (ushort4*)Wb, (int)(nW / 4));
  transpose_convert_v<<<dim3(D / 32, L / 32, B), dim3(32, 8), 0, stream>>>(V, Vt, L, D);

  const float scale = 0.03125f;  // 1/sqrt(1024)

  // S = scale * Q K^T  (masked), bf16
  gemm_wmma<0><<<dim3(L / 128, L / 128, B), 256, 0, stream>>>(
      Qb, Kb, (void*)S, D, L,
      (long long)L * D, (long long)L * D, (long long)L * L,
      scale, qpad, kpad, nullptr);

  // P = softmax(S) row-wise, in place
  softmax_rows<<<B * L, 256, 0, stream>>>(S, L);

  // O = P V  (via Vt rows), bf16
  gemm_wmma<1><<<dim3(E / 128, L / 128, B), 256, 0, stream>>>(
      S, Vt, (void*)Ob, L, E,
      (long long)L * L, (long long)L * D, (long long)L * E,
      1.0f, nullptr, nullptr, nullptr);

  // out = O W^T + b, f32
  gemm_wmma<2><<<dim3(E / 128, (B * L) / 128, 1), 256, 0, stream>>>(
      Ob, Wb, d_out, D, E, 0LL, 0LL, 0LL,
      1.0f, nullptr, nullptr, bias);
}
